// TargetAttention_661424964267
// MI455X (gfx1250) — matrix operand order repaired and checked by reading, not compiled, not run on hardware
//
#include <hip/hip_runtime.h>
#include <hip/hip_bf16.h>
#include <hip/hip_fp16.h>

// ---------------------------------------------------------------------------
// DIN target attention for MI455X (gfx1250, wave32, WMMA f16 16x16x32).
// One workgroup (256 threads = 8 wave32) per batch row b.
// Memory-bound on the 210MB seq stream -> b128 global loads; compute path is
// f16 WMMA with f32 accumulation (fp32 VALU/16x16x4 would be ~8x slower).
// ---------------------------------------------------------------------------

typedef __attribute__((ext_vector_type(16))) _Float16 v16h;
typedef __attribute__((ext_vector_type(2)))  _Float16 v2h;
typedef __attribute__((ext_vector_type(8)))  float    v8f;

static constexpr int B_  = 4096;
static constexpr int S_  = 200;
static constexpr int SP_ = 208;   // S padded to 13*16
static constexpr int D_  = 64;
static constexpr int H1_ = 80;    // 5*16
static constexpr int H1P = 96;    // K-pad for GEMM2 (multiple of 32)
static constexpr int H2_ = 40;
static constexpr int H2P = 48;    // 3*16
static constexpr float NEG_INF_ = -1.0e9f;

// ---- dynamic LDS layout (bytes) -------------------------------------------
static constexpr size_t OFF_SEQ  = 0;                          // f32 [208][64]
static constexpr size_t OFF_DIN  = OFF_SEQ  + SP_*D_*4;        // f16 [208][256]
static constexpr size_t OFF_W1T  = OFF_DIN  + SP_*256*2;       // f16 [80][256]  (w1 transposed)
static constexpr size_t OFF_H1   = OFF_W1T  + H1_*256*2;       // f16 [208][96]
static constexpr size_t OFF_W2T  = OFF_H1   + SP_*H1P*2;       // f16 [48][96]   (w2 transposed, zero-padded)
static constexpr size_t OFF_H2   = OFF_W2T  + H2P*H1P*2;       // f16 [208][48]
static constexpr size_t OFF_B1   = OFF_H2   + SP_*H2P*2;       // f32 [80]
static constexpr size_t OFF_B2   = OFF_B1   + H1_*4;           // f32 [48]
static constexpr size_t OFF_W3   = OFF_B2   + H2P*4;           // f32 [40]
static constexpr size_t OFF_TP   = OFF_W3   + H2_*4;           // f32 [64]  tproj
static constexpr size_t OFF_PW64 = OFF_TP   + D_*4;            // f32 [64]  proj_w row 64
static constexpr size_t OFF_PW65 = OFF_PW64 + D_*4;            // f32 [64]  proj_w row 65
static constexpr size_t OFF_POS  = OFF_PW65 + D_*4;            // f32 [200][2]
static constexpr size_t OFF_SC   = OFF_POS  + S_*2*4;          // f32 [208]  scores -> attn
static constexpr size_t OFF_RED  = OFF_SC   + SP_*4;           // f32 [256]  reduction scratch
static constexpr size_t SMEM_BYTES = OFF_RED + 256*4;          // ~274.7 KB (< 320 KB/WGP)

__device__ __forceinline__ float sigmoidf_(float x) {
    return 1.0f / (1.0f + __expf(-x));
}

// A-fragment per ISA 7.12.2 (16-bit A 16x32): lane = 16*hf + m%16.
// elements e=0..7  -> K = k0 + e + 8*hf
// elements e=8..15 -> K = k0 + 16 + (e-8) + 8*hf
// Two contiguous 16B runs per lane -> compiler emits 2x ds_load_b128.
__device__ __forceinline__ v16h load_a_frag(const _Float16* A, int row, int ldk,
                                            int k0, int hf) {
    v16h a;
    const _Float16* p = A + row * ldk + k0 + 8 * hf;
#pragma unroll
    for (int i = 0; i < 4; ++i) {
        v2h x = *(const v2h*)(p + 2 * i);
        a[2 * i]     = x[0];
        a[2 * i + 1] = x[1];
    }
#pragma unroll
    for (int i = 0; i < 4; ++i) {
        v2h x = *(const v2h*)(p + 16 + 2 * i);
        a[8 + 2 * i] = x[0];
        a[9 + 2 * i] = x[1];
    }
    return a;
}

// Tiled WMMA GEMM with fused bias + sigmoid, f16 in / f32 acc / f16 out.
// A: [MT*16][LDK] row-major f16 (LDS). B stored transposed [NT*16][LDK] f16
// so the B fragment (same K order as A: two 8-wide runs per lane) is two 16B reads.
template <int LDK, int LDN, int MT, int NT, int KC>
__device__ __forceinline__ void gemm_sigmoid(const _Float16* __restrict__ A,
                                             const _Float16* __restrict__ Bt,
                                             const float* __restrict__ bias,
                                             _Float16* __restrict__ O,
                                             int wave, int lane) {
    const int hf  = lane >> 4;
    const int l15 = lane & 15;
    for (int tt = wave; tt < MT * NT; tt += 8) {
        const int mt = tt % MT;
        const int nt = tt / MT;
        const int m  = mt * 16 + l15;
        const int n  = nt * 16 + l15;
        v8f acc = {};
#pragma unroll
        for (int kc = 0; kc < KC; ++kc) {
            v16h a = load_a_frag(A, m, LDK, kc * 32, hf);
            typedef __attribute__((ext_vector_type(8))) _Float16 v8h_;
            const v8h_ b0 = *(const v8h_*)(Bt + n * LDK + kc * 32 + 8 * hf);
            const v8h_ b1 = *(const v8h_*)(Bt + n * LDK + kc * 32 + 16 + 8 * hf);
            v16h b = __builtin_shufflevector(b0, b1, 0, 1, 2, 3, 4, 5, 6, 7, 8, 9, 10, 11, 12, 13, 14, 15);
            acc = __builtin_amdgcn_wmma_f32_16x16x32_f16(
                false, a, false, b, (short)0, acc, false, false);
        }
        const float bn = bias[n];
#pragma unroll
        for (int r = 0; r < 8; ++r) {
            const int row = mt * 16 + r + 8 * hf;   // D layout: m = r + 8*hf
            O[row * LDN + n] = (_Float16)sigmoidf_(acc[r] + bn);
        }
    }
}

__global__ void din_attn_kernel(const float* __restrict__ seq,      // [B,S,D]
                                const float* __restrict__ tgt,      // [B,D]
                                const int*   __restrict__ mask,     // [B,S]
                                const float* __restrict__ pos,      // [S,2]
                                const float* __restrict__ projw,    // [66,64]
                                const float* __restrict__ projb,    // [64]
                                const float* __restrict__ w1,       // [256,80]
                                const float* __restrict__ b1,       // [80]
                                const float* __restrict__ w2,       // [80,40]
                                const float* __restrict__ b2,       // [40]
                                const float* __restrict__ w3,       // [40,1]
                                const float* __restrict__ b3,       // [1]
                                float* __restrict__ out) {          // [B,D]
    extern __shared__ __align__(32) char smem[];
    float*    seqF  = (float*)   (smem + OFF_SEQ);
    _Float16* dinH  = (_Float16*)(smem + OFF_DIN);
    _Float16* w1T   = (_Float16*)(smem + OFF_W1T);
    _Float16* h1H   = (_Float16*)(smem + OFF_H1);
    _Float16* w2T   = (_Float16*)(smem + OFF_W2T);
    _Float16* h2H   = (_Float16*)(smem + OFF_H2);
    float*    b1f   = (float*)   (smem + OFF_B1);
    float*    b2f   = (float*)   (smem + OFF_B2);
    float*    w3f   = (float*)   (smem + OFF_W3);
    float*    tpro  = (float*)   (smem + OFF_TP);
    float*    pw64  = (float*)   (smem + OFF_PW64);
    float*    pw65  = (float*)   (smem + OFF_PW65);
    float*    posL  = (float*)   (smem + OFF_POS);
    float*    score = (float*)   (smem + OFF_SC);
    float*    redf  = (float*)   (smem + OFF_RED);

    const int t    = threadIdx.x;
    const int lane = t & 31;
    const int wave = t >> 5;
    const int b    = blockIdx.x;

    const float* seqG = seq + (size_t)b * S_ * D_;
    const float* tgtG = tgt + (size_t)b * D_;
    const int*   mskG = mask + (size_t)b * S_;

    // ---- phase 0: stage constants / inputs into LDS ------------------------
    for (int i = t; i < H1_; i += 256) b1f[i] = b1[i];
    for (int i = t; i < H2P; i += 256) b2f[i] = (i < H2_) ? b2[i] : 0.0f;
    for (int i = t; i < H2_; i += 256) w3f[i] = w3[i];
    for (int i = t; i < S_ * 2; i += 256) posL[i] = pos[i];
    if (t < D_) {
        // tproj[d] = target_emb[b] . proj_w[0:64, d] + proj_b[d]  (rank-1 fold)
        float a = projb[t];
#pragma unroll 8
        for (int k = 0; k < D_; ++k) a += tgtG[k] * projw[k * D_ + t];
        tpro[t] = a;
        pw64[t] = projw[64 * D_ + t];
        pw65[t] = projw[65 * D_ + t];
    }
    // seq -> LDS f32 via b128 (dominant HBM stream: 210MB total across grid)
    {
        const float4* sg4 = (const float4*)seqG;
        float4*       sf4 = (float4*)seqF;
        for (int i = t; i < (S_ * D_) / 4; i += 256) sf4[i] = sg4[i];
        const float4 z{0.0f, 0.0f, 0.0f, 0.0f};
        for (int i = t; i < ((SP_ - S_) * D_) / 4; i += 256)
            sf4[(S_ * D_) / 4 + i] = z;                    // pad rows 200..207
    }
    // w1 [256][80] -> w1T [80][256] f16 (b128 global reads, 80 % 4 == 0)
    for (int i = t; i < (256 * H1_) / 4; i += 256) {
        const float4 v = ((const float4*)w1)[i];
        const int flat = 4 * i;
        const int k = flat / H1_, n = flat % H1_;
        w1T[(n + 0) * 256 + k] = (_Float16)v.x;
        w1T[(n + 1) * 256 + k] = (_Float16)v.y;
        w1T[(n + 2) * 256 + k] = (_Float16)v.z;
        w1T[(n + 3) * 256 + k] = (_Float16)v.w;
    }
    // w2T zero (covers K pad 80..95 and N pad 40..47)
    for (int i = t; i < H2P * H1P; i += 256) w2T[i] = (_Float16)0.0f;
    // h1 pad columns 80..95 = 0 (so GEMM2's K=96 works)
    for (int i = t; i < SP_ * 16; i += 256) {
        const int s = i >> 4, c = H1_ + (i & 15);
        h1H[s * H1P + c] = (_Float16)0.0f;
    }
    __syncthreads();

    // w2 [80][40] -> w2T [48][96] f16 (after zero fill; 40 % 4 == 0)
    for (int i = t; i < (H1_ * H2_) / 4; i += 256) {
        const float4 v = ((const float4*)w2)[i];
        const int flat = 4 * i;
        const int k = flat / H2_, n = flat % H2_;
        w2T[(n + 0) * H1P + k] = (_Float16)v.x;
        w2T[(n + 1) * H1P + k] = (_Float16)v.y;
        w2T[(n + 2) * H1P + k] = (_Float16)v.z;
        w2T[(n + 3) * H1P + k] = (_Float16)v.w;
    }
    // din f16: [tgt | seq | tgt-seq | tgt*seq], tgt = tproj + pos·projw[64:66]
    // Each thread handles a d-pair -> 4x v2h (4B) LDS stores per iter.
    for (int i = t; i < SP_ * (D_ / 2); i += 256) {
        const int s = i >> 5;
        const int d = (i & 31) * 2;
        float tv0 = 0.0f, tv1 = 0.0f, sv0 = 0.0f, sv1 = 0.0f;
        if (s < S_) {
            const float p0 = posL[2 * s], p1 = posL[2 * s + 1];
            tv0 = tpro[d]     + p0 * pw64[d]     + p1 * pw65[d];
            tv1 = tpro[d + 1] + p0 * pw64[d + 1] + p1 * pw65[d + 1];
            sv0 = seqF[s * D_ + d];
            sv1 = seqF[s * D_ + d + 1];
        }
        _Float16* row = dinH + (size_t)s * 256;
        v2h x;
        x[0] = (_Float16)tv0;         x[1] = (_Float16)tv1;
        *(v2h*)(row + d) = x;
        x[0] = (_Float16)sv0;         x[1] = (_Float16)sv1;
        *(v2h*)(row + 64 + d) = x;
        x[0] = (_Float16)(tv0 - sv0); x[1] = (_Float16)(tv1 - sv1);
        *(v2h*)(row + 128 + d) = x;
        x[0] = (_Float16)(tv0 * sv0); x[1] = (_Float16)(tv1 * sv1);
        *(v2h*)(row + 192 + d) = x;
    }
    __syncthreads();

    // ---- GEMM1: [208x256] x [256x80] + b1, sigmoid -> h1 [208x96] ----------
    gemm_sigmoid<256, H1P, 13, 5, 8>(dinH, w1T, b1f, h1H, wave, lane);
    __syncthreads();

    // ---- GEMM2: [208x96] x [96x48] + b2, sigmoid -> h2 [208x48] ------------
    gemm_sigmoid<H1P, H2P, 13, 3, 3>(h1H, w2T, b2f, h2H, wave, lane);
    __syncthreads();

    // ---- scores: h2 . w3 + b3, masked --------------------------------------
    if (t < S_) {
        float a = b3[0];
        const _Float16* hr = h2H + t * H2P;
#pragma unroll 8
        for (int k = 0; k < H2_; ++k) a += (float)hr[k] * w3f[k];
        score[t] = (mskG[t] == 0) ? NEG_INF_ : a;
    }
    __syncthreads();

    // ---- softmax over s<200 -------------------------------------------------
    redf[t] = (t < S_) ? score[t] : NEG_INF_;
    __syncthreads();
    for (int off = 128; off > 0; off >>= 1) {
        if (t < off) redf[t] = fmaxf(redf[t], redf[t + off]);
        __syncthreads();
    }
    const float mx = redf[0];
    __syncthreads();
    float e = 0.0f;
    if (t < S_) e = __expf(score[t] - mx);
    redf[t] = e;
    if (t < S_) score[t] = e;
    __syncthreads();
    for (int off = 128; off > 0; off >>= 1) {
        if (t < off) redf[t] += redf[t + off];
        __syncthreads();
    }
    const float inv = 1.0f / redf[0];
    __syncthreads();
    if (t < S_) score[t] *= inv;          // attn weights
    __syncthreads();

    // ---- pooling: out[b,d] = sum_s attn[s] * seq[b,s,d] (f32 from LDS) -----
    const int d = t & 63, part = t >> 6;  // 4 partial slices of 50
    float ps = 0.0f;
    for (int s = part * 50; s < part * 50 + 50; ++s)
        ps += score[s] * seqF[s * D_ + d];
    redf[t] = ps;
    __syncthreads();
    if (t < D_) {
        out[(size_t)b * D_ + t] =
            redf[t] + redf[64 + t] + redf[128 + t] + redf[192 + t];
    }
}

extern "C" void kernel_launch(void* const* d_in, const int* in_sizes, int n_in,
                              void* d_out, int out_size, void* d_ws, size_t ws_size,
                              hipStream_t stream) {
    const float* seq   = (const float*)d_in[0];
    const float* tgt   = (const float*)d_in[1];
    const int*   mask  = (const int*)  d_in[2];
    const float* pos   = (const float*)d_in[3];
    const float* projw = (const float*)d_in[4];
    const float* projb = (const float*)d_in[5];
    const float* w1    = (const float*)d_in[6];
    const float* b1    = (const float*)d_in[7];
    const float* w2    = (const float*)d_in[8];
    const float* b2    = (const float*)d_in[9];
    const float* w3    = (const float*)d_in[10];
    const float* b3    = (const float*)d_in[11];
    float* out = (float*)d_out;

    (void)in_sizes; (void)n_in; (void)out_size; (void)d_ws; (void)ws_size;

    // Allow >64KB dynamic LDS (CDNA5: up to 320KB per workgroup).
    static_assert(SMEM_BYTES <= 320 * 1024, "LDS budget exceeded");
    hipFuncSetAttribute((const void*)din_attn_kernel,
                        hipFuncAttributeMaxDynamicSharedMemorySize,
                        (int)SMEM_BYTES);

    din_attn_kernel<<<B_, 256, SMEM_BYTES, stream>>>(
        seq, tgt, mask, pos, projw, projb, w1, b1, w2, b2, w3, b3, out);
}
